// CRFLayer_10969346474701
// MI455X (gfx1250) — matrix operand order repaired and checked by reading, not compiled, not run on hardware
//
#include <hip/hip_runtime.h>

// CRF forward (partition function) SEQ=512, BATCH=1024, TAGS=32.
// Per wave (wave32): 16 batch columns. alpha kept in WMMA D-layout f32 regs,
// in log2 domain (alpha2 = alpha * log2(e)) so the per-step chain is:
//   m = max(alpha2) ; u = exp2(alpha2 - m) packed f16 -> B layout via
//   v_permlane16_swap ; v = E @ u + 1e-37 via 2x v_wmma_f32_16x16x32_f16
//   (the 1e-37 floor rides in the WMMA C accumulator: all products are >= 0,
//    so D >= 1e-37 by construction -> v_log_f32 consumes D directly) ;
//   alpha2' = fma(feat, log2e, m) + log2(v).
// E = exp(transition) precomputed once in f16 A-layout. feats(t+1) register-
// prefetched during step t. HBM floor: 64 MB feats streamed once (~2.8 us).

typedef __attribute__((ext_vector_type(16))) _Float16 v16h;
typedef __attribute__((ext_vector_type(8)))  float    v8f;
typedef __attribute__((ext_vector_type(2)))  __fp16   v2hf;
typedef __attribute__((ext_vector_type(2)))  unsigned v2u;

#define SEQ_LEN  512
#define BATCHN   1024
#define TAGS     32
#define END_IDX  31
#define NEGV     (-10000.0f)
#define L2E      1.44269504088896340736f   // log2(e)
#define LN2      0.69314718055994530942f
#define VFLOOR   1.0e-37f                  // > min normal f32; WMMA C bias

__device__ __forceinline__ unsigned pk2u(float x, float y) {
  union { v2hf h; unsigned u; } c;
  c.h = __builtin_amdgcn_cvt_pkrtz(x, y);   // v_cvt_pk_rtz_f16_f32
  return c.u;
}

__device__ __forceinline__ float ex2(float x) {   // 2^x, raw v_exp_f32
#if __has_builtin(__builtin_amdgcn_exp2f)
  return __builtin_amdgcn_exp2f(x);
#else
  return exp2f(x);
#endif
}
__device__ __forceinline__ float lg2(float x) {   // log2(x), raw v_log_f32
#if __has_builtin(__builtin_amdgcn_logf)
  return __builtin_amdgcn_logf(x);
#else
  return __log2f(x);
#endif
}

// Merge two D-layout packed-f16 words (a = rows i0..i0+7 pairs, b = +16 tile)
// into the two B-layout words: out_lo = {a.lo | b.lo}, out_hi = {a.hi | b.hi}.
__device__ __forceinline__ void bmerge(unsigned a, unsigned b, bool lo,
                                       unsigned& out_lo, unsigned& out_hi) {
#if __has_builtin(__builtin_amdgcn_permlane16_swap)
  // v_permlane16_swap_b32: swaps lanes 0-15 of vdst with lanes 16-31 of vsrc.
  // With (vdst=b, vsrc=a):  new_vdst = {a.hi, b.hi} ; new_vsrc = {a.lo, b.lo}.
  v2u r = __builtin_amdgcn_permlane16_swap(b, a, false, false);
  out_hi = r[0];
  out_lo = r[1];
  (void)lo;
#else
  unsigned swa = (unsigned)__shfl_xor((int)a, 16, 32);
  unsigned swb = (unsigned)__shfl_xor((int)b, 16, 32);
  out_lo = lo ? a   : swb;
  out_hi = lo ? swa : b;
#endif
}

// Value of x held by the opposite 16-lane half (per-batch reduce combine).
__device__ __forceinline__ float xhalf(float x, bool lo) {
#if __has_builtin(__builtin_amdgcn_permlane16_swap)
  union { float f; unsigned u; } c; c.f = x;
  v2u r = __builtin_amdgcn_permlane16_swap(c.u, c.u, false, false);
  // r[0] lanes0-15 = x.hi ; r[1] lanes16-31 = x.lo
  union { unsigned u; float f; } o; o.u = lo ? r[0] : r[1];
  return o.f;
#else
  return __shfl_xor(x, 16, 32);
#endif
}

__global__ __launch_bounds__(128)
void crf_fwd_wmma(const float* __restrict__ feats,
                  const float* __restrict__ mask,
                  const float* __restrict__ trans,
                  float* __restrict__ out)
{
  const int lane = threadIdx.x & 31;
  const int wid  = (int)((blockIdx.x * blockDim.x + threadIdx.x) >> 5);
  const int B0   = wid * 16;                 // 16 batches per wave
  const bool lo  = lane < 16;
  const int bsel = lane & 15;                // batch column owned by this lane
  const int i0   = lo ? 0 : 8;               // D-layout row offset of this half

  // ---- A operands: E = exp(transition) in f16 A-layout (M = lane%16) ----
  //   K = 2p+h (+16 for lanes>=16)
  union { unsigned u[8]; v16h v; } UA0, UA1;
#pragma unroll
  for (int p = 0; p < 8; ++p) {
    int kb = 2 * p;
    int k0 = kb + (lo ? 0 : 16);
    float e00 = ex2(L2E * trans[(bsel     ) * TAGS + k0    ]);
    float e01 = ex2(L2E * trans[(bsel     ) * TAGS + k0 + 1]);
    float e10 = ex2(L2E * trans[(bsel + 16) * TAGS + k0    ]);
    float e11 = ex2(L2E * trans[(bsel + 16) * TAGS + k0 + 1]);
    UA0.u[p] = pk2u(e00, e01);
    UA1.u[p] = pk2u(e10, e11);
  }
  const v16h A0 = UA0.v;
  const v16h A1 = UA1.v;

  // WMMA C accumulator = 1e-37 floor (keeps D normal and log-safe)
  const v8f cb = { VFLOOR, VFLOOR, VFLOOR, VFLOOR,
                   VFLOOR, VFLOOR, VFLOOR, VFLOOR };

  // ---- alpha2 (log2 domain) in D layout: aa = tile0 (i=i0+r), ab = tile1 ----
  float aa[8], ab[8];
#pragma unroll
  for (int r = 0; r < 8; ++r) { aa[r] = NEGV * L2E; ab[r] = NEGV * L2E; }
  if (!lo) ab[6] = 0.0f;                     // i = 30 == START_IDX

  // ---- prefetch feats(t=0), mask(t=0) ----
  float4 nc0, nc1, nc2, nc3; float nmk;
  {
    const float* fp = feats + ((size_t)(B0 + bsel)) * TAGS + i0;
    nc0 = *(const float4*)(fp +  0);
    nc1 = *(const float4*)(fp +  4);
    nc2 = *(const float4*)(fp + 16);
    nc3 = *(const float4*)(fp + 20);
    nmk = mask[B0 + bsel];
  }

  for (int t = 0; t < SEQ_LEN; ++t) {
    float4 c0 = nc0, c1 = nc1, c2 = nc2, c3 = nc3;
    float mk = nmk;

    // prefetch next timestep (register double-buffer; off critical path)
    int tn = (t + 1 < SEQ_LEN) ? (t + 1) : t;
    const float* fp = feats + ((size_t)tn * BATCHN + B0 + bsel) * TAGS + i0;
    nc0 = *(const float4*)(fp +  0);
    nc1 = *(const float4*)(fp +  4);
    nc2 = *(const float4*)(fp + 16);
    nc3 = *(const float4*)(fp + 20);
    nmk = mask[(size_t)tn * BATCHN + B0 + bsel];

    // ---- per-batch running max over all 32 tags (log2 domain) ----
    float x0 = fmaxf(fmaxf(aa[0], aa[1]), fmaxf(aa[2], aa[3]));
    float x1 = fmaxf(fmaxf(aa[4], aa[5]), fmaxf(aa[6], aa[7]));
    float y0 = fmaxf(fmaxf(ab[0], ab[1]), fmaxf(ab[2], ab[3]));
    float y1 = fmaxf(fmaxf(ab[4], ab[5]), fmaxf(ab[6], ab[7]));
    float m  = fmaxf(fmaxf(x0, x1), fmaxf(y0, y1));
    m = fmaxf(m, xhalf(m, lo));

    // ---- u = exp2(alpha2 - m), packed f16 pairs, D order ----
    unsigned pka[4], pkb[4];
#pragma unroll
    for (int p = 0; p < 4; ++p) {
      pka[p] = pk2u(ex2(aa[2 * p] - m), ex2(aa[2 * p + 1] - m));
      pkb[p] = pk2u(ex2(ab[2 * p] - m), ex2(ab[2 * p + 1] - m));
    }
    // ---- D-layout -> B-layout (one permlane16_swap per pair) ----
    union { unsigned u[8]; v16h v; } U;
#pragma unroll
    for (int p = 0; p < 4; ++p)
      bmerge(pka[p], pkb[p], lo, U.u[p], U.u[p + 4]);

    // ---- v = E @ u + 1e-37 : two WMMAs cover i = 0..15 and i = 16..31 ----
    v8f d0 = __builtin_amdgcn_wmma_f32_16x16x32_f16(
        false, A0, false, U.v, (short)0, cb, false, false);
    v8f d1 = __builtin_amdgcn_wmma_f32_16x16x32_f16(
        false, A1, false, U.v, (short)0, cb, false, false);

    // ---- alpha2' = fma(feat, log2e, m) + log2(v), masked select ----
    float fa[8] = { c0.x, c0.y, c0.z, c0.w, c1.x, c1.y, c1.z, c1.w };
    float fb[8] = { c2.x, c2.y, c2.z, c2.w, c3.x, c3.y, c3.z, c3.w };
    bool upd = mk > 0.5f;
#pragma unroll
    for (int r = 0; r < 8; ++r) {
      float ba = fmaf(fa[r], L2E, m);          // ready before WMMA completes
      float bb = fmaf(fb[r], L2E, m);
      float na = ba + lg2(d0[r]);              // D >= 1e-37, no clamp needed
      float nb = bb + lg2(d1[r]);
      aa[r] = upd ? na : aa[r];
      ab[r] = upd ? nb : ab[r];
    }
  }

  // ---- finalize: alpha2 += log2e*T[END][i]; out = LN2*(m2 + log2(sum)) ----
  const float* te = trans + END_IDX * TAGS;
#pragma unroll
  for (int r = 0; r < 8; ++r) {
    aa[r] = fmaf(te[i0 + r],      L2E, aa[r]);
    ab[r] = fmaf(te[16 + i0 + r], L2E, ab[r]);
  }
  float x0 = fmaxf(fmaxf(aa[0], aa[1]), fmaxf(aa[2], aa[3]));
  float x1 = fmaxf(fmaxf(aa[4], aa[5]), fmaxf(aa[6], aa[7]));
  float y0 = fmaxf(fmaxf(ab[0], ab[1]), fmaxf(ab[2], ab[3]));
  float y1 = fmaxf(fmaxf(ab[4], ab[5]), fmaxf(ab[6], ab[7]));
  float m2 = fmaxf(fmaxf(x0, x1), fmaxf(y0, y1));
  m2 = fmaxf(m2, xhalf(m2, lo));
  float s = 0.0f;
#pragma unroll
  for (int r = 0; r < 8; ++r)
    s += ex2(aa[r] - m2) + ex2(ab[r] - m2);
  s += xhalf(s, lo);
  if (lo) out[B0 + bsel] = LN2 * (m2 + lg2(s));
}

extern "C" void kernel_launch(void* const* d_in, const int* in_sizes, int n_in,
                              void* d_out, int out_size, void* d_ws, size_t ws_size,
                              hipStream_t stream) {
  (void)in_sizes; (void)n_in; (void)d_ws; (void)ws_size; (void)out_size;
  const float* feats = (const float*)d_in[0];   // (512, 1024, 32) f32
  const float* mask  = (const float*)d_in[1];   // (512, 1024)     f32
  const float* trans = (const float*)d_in[2];   // (32, 32)        f32
  float* out = (float*)d_out;                   // (1024,)         f32

  // 1024 batches / 16 per wave = 64 waves: 16 blocks x 128 threads (4 waves/WG)
  crf_fwd_wmma<<<dim3(16), dim3(128), 0, stream>>>(feats, mask, trans, out);
}
